// MultiHeadAttention_53790170415152
// MI455X (gfx1250) — matrix operand order repaired and checked by reading, not compiled, not run on hardware
//
#include <hip/hip_runtime.h>
#include <hip/hip_bf16.h>

// ---------------- problem constants (match reference) ----------------
#define Bn 2
#define Tn 2048
#define Dn 1024
#define Hn 16
#define HDn 64
#define Mn (Bn * Tn) // 4096 rows for the projections

typedef _Float16 half_t;
typedef half_t v8h  __attribute__((ext_vector_type(8)));
typedef half_t v16h __attribute__((ext_vector_type(16)));
typedef float  v8f  __attribute__((ext_vector_type(8)));

union Frag16 { v16h v; v8h h[2]; };

// ---------------- WMMA fragment loaders (wave32, 16x16x32 f16) ----------------
// A (16x32, MxK): lane holds row M = lane%16; 16 contiguous K values starting at
// (lane/16)*16 (matches B fragment order).
__device__ __forceinline__ v16h load_frag_a(const half_t* __restrict__ base, int ld, int kbase) {
  const int lane = threadIdx.x & 31;
  const int row  = lane & 15;
  const int ko   = (lane >> 4) << 4;
  const half_t* p = base + (size_t)row * ld + kbase;
  Frag16 f;
  f.h[0] = *(const v8h*)(p + ko);
  f.h[1] = *(const v8h*)(p + ko + 8);
  return f.v;
}
// B (32x16, KxN): lane holds col N = lane%16; 16 contiguous K values starting at
// (lane/16)*16 (same K order as the A fragment).
__device__ __forceinline__ v16h load_frag_b(const half_t* __restrict__ base, int ld, int kbase) {
  const int lane = threadIdx.x & 31;
  const int n    = lane & 15;
  const int ko   = (lane >> 4) << 4;
  const half_t* p = base + (size_t)n * ld + kbase + ko;
  Frag16 f;
  f.h[0] = *(const v8h*)(p);
  f.h[1] = *(const v8h*)(p + 8);
  return f.v;
}

__device__ __forceinline__ v8f wmma_f16(v16h a, v16h b, v8f c) {
  return __builtin_amdgcn_wmma_f32_16x16x32_f16(false, a, false, b, (short)0, c, false, false);
}

__device__ __forceinline__ v8f v8f_zero() {
  v8f z = {0.f, 0.f, 0.f, 0.f, 0.f, 0.f, 0.f, 0.f};
  return z;
}

// ---------------- kernel 0: f32 -> f16 convert (grid-stride) ----------------
__global__ void cvt_f32_to_f16(const float* __restrict__ in, half_t* __restrict__ out, int n) {
  for (int i = blockIdx.x * blockDim.x + threadIdx.x; i < n; i += gridDim.x * blockDim.x)
    out[i] = (half_t)in[i];
}

// ---------------- kernel 1: gate scalars g[b,h,t] ----------------
__global__ __launch_bounds__(256) void gates_kernel(const float* __restrict__ x,
                                                    const float* __restrict__ Wg,
                                                    const float* __restrict__ bg,
                                                    const float* __restrict__ gru,
                                                    float* __restrict__ g) {
  __shared__ float wgs[8][64];
  __shared__ float bgs[8];
  for (int i = threadIdx.x; i < 512; i += 256) wgs[i >> 6][i & 63] = Wg[i];
  if (threadIdx.x < 8) bgs[threadIdx.x] = bg[threadIdx.x];
  __syncthreads();

  const int idx = blockIdx.x * 256 + threadIdx.x; // idx = b*H*T + h*T + t
  const int t = idx & (Tn - 1);
  const int h = (idx >> 11) & (Hn - 1);
  const int b = idx >> 15;
  const float* xp = x + ((size_t)b * Tn + t) * Dn + h * HDn;

  float acc[8];
#pragma unroll
  for (int e = 0; e < 8; ++e) acc[e] = 0.f;
  for (int d = 0; d < HDn; ++d) {
    float xv = xp[d];
#pragma unroll
    for (int e = 0; e < 8; ++e) acc[e] += xv * wgs[e][d];
  }
  float sA = bgs[0] + bgs[1] + bgs[2] + bgs[3];
  float sB = bgs[4] + bgs[5] + bgs[6] + bgs[7];
#pragma unroll
  for (int e = 0; e < 4; ++e) sA += acc[e];
#pragma unroll
  for (int e = 4; e < 8; ++e) sB += acc[e];
  const float ga = 1.f / (1.f + __expf(-sA));
  const float gb = 1.f / (1.f + __expf(-sB));
  g[idx] = ga * (gb * gru[0] - 1.0f) + 2.0f;
}

// ---------------- kernel 2: QKV projections (f16 WMMA, ping-pong pipelined) ----------------
// grid: (Mn/128, Dn/64, 3). Each wave computes a 16x64 output strip. The k-loop
// is unrolled 2x so the two fragment buffers alternate roles structurally:
// no inter-buffer v_mov copies, loads stay in flight across the WMMAs.
__global__ __launch_bounds__(256) void qkv_gemm(const half_t* __restrict__ x,
                                                const half_t* __restrict__ Wq,
                                                const half_t* __restrict__ Wk,
                                                const half_t* __restrict__ Wv,
                                                const float* __restrict__ bqv,
                                                const float* __restrict__ bkv,
                                                const float* __restrict__ bvv,
                                                half_t* __restrict__ qo,
                                                half_t* __restrict__ ko,
                                                half_t* __restrict__ vto) {
  const int which = blockIdx.z;
  const half_t* W   = (which == 0) ? Wq : (which == 1) ? Wk : Wv;
  const float* bias = (which == 0) ? bqv : (which == 1) ? bkv : bvv;

  const int wid  = threadIdx.x >> 5;
  const int lane = threadIdx.x & 31;
  const int m0 = blockIdx.x * 128 + wid * 16;
  const int n0 = blockIdx.y * 64;

  v8f acc[4];
#pragma unroll
  for (int j = 0; j < 4; ++j) acc[j] = v8f_zero();

  const half_t* xa = x + (size_t)m0 * Dn;
  v16h a0 = load_frag_a(xa, Dn, 0);
  v16h b0[4];
#pragma unroll
  for (int j = 0; j < 4; ++j) b0[j] = load_frag_b(W + (size_t)(n0 + j * 16) * Dn, Dn, 0);

  for (int k0 = 0; k0 < Dn - 64; k0 += 64) {
    v16h a1 = load_frag_a(xa, Dn, k0 + 32);
    v16h b1[4];
#pragma unroll
    for (int j = 0; j < 4; ++j)
      b1[j] = load_frag_b(W + (size_t)(n0 + j * 16) * Dn, Dn, k0 + 32);
#pragma unroll
    for (int j = 0; j < 4; ++j) acc[j] = wmma_f16(a0, b0[j], acc[j]);

    a0 = load_frag_a(xa, Dn, k0 + 64);
#pragma unroll
    for (int j = 0; j < 4; ++j)
      b0[j] = load_frag_b(W + (size_t)(n0 + j * 16) * Dn, Dn, k0 + 64);
#pragma unroll
    for (int j = 0; j < 4; ++j) acc[j] = wmma_f16(a1, b1[j], acc[j]);
  }
  { // tail: k-steps Dn-64 (in a0/b0) and Dn-32
    v16h a1 = load_frag_a(xa, Dn, Dn - 32);
    v16h b1[4];
#pragma unroll
    for (int j = 0; j < 4; ++j)
      b1[j] = load_frag_b(W + (size_t)(n0 + j * 16) * Dn, Dn, Dn - 32);
#pragma unroll
    for (int j = 0; j < 4; ++j) acc[j] = wmma_f16(a0, b0[j], acc[j]);
#pragma unroll
    for (int j = 0; j < 4; ++j) acc[j] = wmma_f16(a1, b1[j], acc[j]);
  }

  const int mhi = (lane >> 4) << 3;
  const int nl  = lane & 15;
#pragma unroll
  for (int j = 0; j < 4; ++j) {
    const int n = n0 + j * 16 + nl;
    const int h = n >> 6, d = n & 63;
    const float bb = bias[n];
#pragma unroll
    for (int r = 0; r < 8; ++r) {
      const int m = m0 + r + mhi;
      const int b = m >> 11, t = m & (Tn - 1);
      const half_t val = (half_t)(acc[j][r] + bb);
      if (which == 2)
        vto[(((size_t)b * Hn + h) * HDn + d) * Tn + t] = val;
      else if (which == 1)
        ko[(((size_t)b * Hn + h) * Tn + t) * HDn + d] = val;
      else
        qo[(((size_t)b * Hn + h) * Tn + t) * HDn + d] = val;
    }
  }
}

// ---------------- kernel 3: fused flash attention with gated bias ----------------
// grid: (H * T/128, B), 256 threads (8 waves). All waves share (b,h): K/V key
// tiles are staged cooperatively into LDS once per iteration (8x fewer global
// loads), waves read WMMA fragments via ds_load_b128. Bias streamed to regs.
__global__ __launch_bounds__(256) void attn_fused(const half_t* __restrict__ q,
                                                  const half_t* __restrict__ k,
                                                  const half_t* __restrict__ vt,
                                                  const float* __restrict__ pbias,
                                                  const float* __restrict__ g,
                                                  half_t* __restrict__ out) {
  __shared__ __align__(16) half_t K_lds[64][72];      // 64 keys x 64 d (+pad)
  __shared__ __align__(16) half_t V_lds[64][72];      // 64 d x 64 keys (+pad)
  __shared__ __align__(16) half_t p_lds[8][16][72];   // per-wave P staging

  const int tid  = threadIdx.x;
  const int wid  = tid >> 5;
  const int lane = tid & 31;
  const int b  = blockIdx.y;
  const int h  = blockIdx.x >> 4;       // T/128 == 16 q-tiles per head
  const int qt = blockIdx.x & 15;
  const int q0 = qt * 128 + wid * 16;

  const size_t bh = (size_t)b * Hn + h;
  const half_t* qbase = q  + (bh * Tn + q0) * HDn;
  const half_t* kbase = k  + bh * Tn * HDn;
  const half_t* vbase = vt + bh * HDn * Tn;
  const float*  pb    = pbias + ((size_t)h * Tn + q0) * Tn;
  const float*  gb    = g + bh * Tn + q0;

  const v16h qa0 = load_frag_a(qbase, HDn, 0);
  const v16h qa1 = load_frag_a(qbase, HDn, 32);

  const int mhi = (lane >> 4) << 3;
  const int nl  = lane & 15;
  const int ko  = (lane >> 4) << 4;     // B-fragment K offset
  float gr[8], mi[8], li[8];
#pragma unroll
  for (int r = 0; r < 8; ++r) { gr[r] = gb[r + mhi]; mi[r] = -1e30f; li[r] = 0.f; }
  v8f o[4];
#pragma unroll
  for (int j = 0; j < 4; ++j) o[j] = v8f_zero();

  const float scale = 0.125f; // 1/sqrt(64)

  // cooperative staging pattern: 256 threads x 32B cover one 64x64 f16 tile
  const int srow   = tid >> 2;          // 0..63
  const int schunk = (tid & 3) * 16;    // 0,16,32,48

  for (int key0 = 0; key0 < Tn; key0 += 64) {
    // ---- stage K (keys x d) and V (d x keys) tiles into LDS ----
    {
      const half_t* kg = kbase + (size_t)(key0 + srow) * HDn + schunk;
      *(v8h*)&K_lds[srow][schunk]     = *(const v8h*)kg;
      *(v8h*)&K_lds[srow][schunk + 8] = *(const v8h*)(kg + 8);
      const half_t* vg = vbase + (size_t)srow * Tn + key0 + schunk;
      *(v8h*)&V_lds[srow][schunk]     = *(const v8h*)vg;
      *(v8h*)&V_lds[srow][schunk + 8] = *(const v8h*)(vg + 8);
    }
    if (key0 + 64 < Tn)
      __builtin_prefetch(pb + (size_t)mhi * Tn + key0 + 64, 0, 1); // global_prefetch_b8
    __syncthreads();

    // ---- bias loads first: in flight during the S-tile WMMAs ----
    float pbv[4][8];
#pragma unroll
    for (int j = 0; j < 4; ++j)
#pragma unroll
      for (int r = 0; r < 8; ++r)
        pbv[j][r] = pb[(size_t)(r + mhi) * Tn + key0 + j * 16 + nl];

    // ---- S tile: 16 q-rows x 64 keys (K fragments from LDS) ----
    v8f s[4];
#pragma unroll
    for (int j = 0; j < 4; ++j) {
      Frag16 b0, b1;
      b0.h[0] = *(const v8h*)&K_lds[j * 16 + nl][ko];
      b0.h[1] = *(const v8h*)&K_lds[j * 16 + nl][ko + 8];
      b1.h[0] = *(const v8h*)&K_lds[j * 16 + nl][32 + ko];
      b1.h[1] = *(const v8h*)&K_lds[j * 16 + nl][32 + ko + 8];
      v8f z = v8f_zero();
      z = wmma_f16(qa0, b0.v, z);
      z = wmma_f16(qa1, b1.v, z);
      s[j] = z;
    }

    // ---- scale + gated position bias ----
#pragma unroll
    for (int j = 0; j < 4; ++j)
#pragma unroll
      for (int r = 0; r < 8; ++r)
        s[j][r] = s[j][r] * scale + gr[r] * pbv[j][r];

    // ---- online softmax: row max over 64 keys (4 tiles x 16 lanes) ----
    float tmax[8];
#pragma unroll
    for (int r = 0; r < 8; ++r)
      tmax[r] = fmaxf(fmaxf(s[0][r], s[1][r]), fmaxf(s[2][r], s[3][r]));
#pragma unroll
    for (int off = 8; off; off >>= 1)
#pragma unroll
      for (int r = 0; r < 8; ++r)
        tmax[r] = fmaxf(tmax[r], __shfl_xor(tmax[r], off, 32));

    float al[8], tsum[8];
#pragma unroll
    for (int r = 0; r < 8; ++r) {
      const float nm = fmaxf(mi[r], tmax[r]);
      al[r] = __expf(mi[r] - nm);
      mi[r] = nm;
      tsum[r] = 0.f;
    }
#pragma unroll
    for (int j = 0; j < 4; ++j)
#pragma unroll
      for (int r = 0; r < 8; ++r) {
        const float e = __expf(s[j][r] - mi[r]);
        s[j][r] = e;
        tsum[r] += e;
      }
#pragma unroll
    for (int off = 8; off; off >>= 1)
#pragma unroll
      for (int r = 0; r < 8; ++r)
        tsum[r] += __shfl_xor(tsum[r], off, 32);
#pragma unroll
    for (int r = 0; r < 8; ++r) li[r] = li[r] * al[r] + tsum[r];
#pragma unroll
    for (int j = 0; j < 4; ++j)
#pragma unroll
      for (int r = 0; r < 8; ++r) o[j][r] *= al[r];

    // ---- P: C-layout -> f16 A-layout via wave-private LDS ----
#pragma unroll
    for (int j = 0; j < 4; ++j)
#pragma unroll
      for (int r = 0; r < 8; ++r)
        p_lds[wid][r + mhi][j * 16 + nl] = (half_t)s[j][r];

    // ---- O += P @ V  (P and V fragments from LDS) ----
#pragma unroll
    for (int ks = 0; ks < 64; ks += 32) {
      Frag16 pf;
      const int prow = lane & 15;
      const int pko  = (lane >> 4) << 4;
      pf.h[0] = *(const v8h*)&p_lds[wid][prow][ks + pko];
      pf.h[1] = *(const v8h*)&p_lds[wid][prow][ks + pko + 8];
#pragma unroll
      for (int dj = 0; dj < 4; ++dj) {
        Frag16 vf;
        vf.h[0] = *(const v8h*)&V_lds[dj * 16 + nl][ks + ko];
        vf.h[1] = *(const v8h*)&V_lds[dj * 16 + nl][ks + ko + 8];
        o[dj] = wmma_f16(pf.v, vf.v, o[dj]);
      }
    }
    __syncthreads(); // K/V tiles fully consumed before next staging pass
  }

  // ---- normalize and write f16 for the output projection: out[b,t,h*64+d] ----
#pragma unroll
  for (int dj = 0; dj < 4; ++dj) {
    const int n = h * HDn + dj * 16 + nl;
#pragma unroll
    for (int r = 0; r < 8; ++r) {
      const int t = q0 + r + mhi;
      out[((size_t)b * Tn + t) * Dn + n] = (half_t)(o[dj][r] / li[r]);
    }
  }
}

// ---------------- kernel 4: output projection (f16 WMMA, ping-pong pipelined) ----------------
__global__ __launch_bounds__(256) void out_gemm(const half_t* __restrict__ a,
                                                const half_t* __restrict__ Wo,
                                                const float* __restrict__ bo,
                                                float* __restrict__ out) {
  const int wid  = threadIdx.x >> 5;
  const int lane = threadIdx.x & 31;
  const int m0 = blockIdx.x * 128 + wid * 16;
  const int n0 = blockIdx.y * 64;

  v8f acc[4];
#pragma unroll
  for (int j = 0; j < 4; ++j) acc[j] = v8f_zero();

  const half_t* xa = a + (size_t)m0 * Dn;
  v16h a0 = load_frag_a(xa, Dn, 0);
  v16h b0[4];
#pragma unroll
  for (int j = 0; j < 4; ++j) b0[j] = load_frag_b(Wo + (size_t)(n0 + j * 16) * Dn, Dn, 0);

  for (int k0 = 0; k0 < Dn - 64; k0 += 64) {
    v16h a1 = load_frag_a(xa, Dn, k0 + 32);
    v16h b1[4];
#pragma unroll
    for (int j = 0; j < 4; ++j)
      b1[j] = load_frag_b(Wo + (size_t)(n0 + j * 16) * Dn, Dn, k0 + 32);
#pragma unroll
    for (int j = 0; j < 4; ++j) acc[j] = wmma_f16(a0, b0[j], acc[j]);

    a0 = load_frag_a(xa, Dn, k0 + 64);
#pragma unroll
    for (int j = 0; j < 4; ++j)
      b0[j] = load_frag_b(Wo + (size_t)(n0 + j * 16) * Dn, Dn, k0 + 64);
#pragma unroll
    for (int j = 0; j < 4; ++j) acc[j] = wmma_f16(a1, b1[j], acc[j]);
  }
  { // tail
    v16h a1 = load_frag_a(xa, Dn, Dn - 32);
    v16h b1[4];
#pragma unroll
    for (int j = 0; j < 4; ++j)
      b1[j] = load_frag_b(Wo + (size_t)(n0 + j * 16) * Dn, Dn, Dn - 32);
#pragma unroll
    for (int j = 0; j < 4; ++j) acc[j] = wmma_f16(a0, b0[j], acc[j]);
#pragma unroll
    for (int j = 0; j < 4; ++j) acc[j] = wmma_f16(a1, b1[j], acc[j]);
  }

  const int mhi = (lane >> 4) << 3;
  const int nl  = lane & 15;
#pragma unroll
  for (int j = 0; j < 4; ++j) {
    const int n = n0 + j * 16 + nl;
    const float bb = bo[n];
#pragma unroll
    for (int r = 0; r < 8; ++r)
      out[(size_t)(m0 + r + mhi) * Dn + n] = acc[j][r] + bb;
  }
}

// ---------------- host-side launch ----------------
extern "C" void kernel_launch(void* const* d_in, const int* in_sizes, int n_in,
                              void* d_out, int out_size, void* d_ws, size_t ws_size,
                              hipStream_t stream) {
  (void)in_sizes; (void)n_in; (void)out_size; (void)ws_size;
  const float* x     = (const float*)d_in[0];
  const float* pbias = (const float*)d_in[1];
  const float* Wq    = (const float*)d_in[2];
  const float* bq    = (const float*)d_in[3];
  const float* Wk    = (const float*)d_in[4];
  const float* bk    = (const float*)d_in[5];
  const float* Wv    = (const float*)d_in[6];
  const float* bv    = (const float*)d_in[7];
  const float* Wo    = (const float*)d_in[8];
  const float* bo    = (const float*)d_in[9];
  const float* Wg    = (const float*)d_in[10];
  const float* bg    = (const float*)d_in[11];
  const float* gru   = (const float*)d_in[12];
  float* out = (float*)d_out;

  char* ws = (char*)d_ws;
  const size_t MB = 1024u * 1024u;
  half_t* xh  = (half_t*)(ws + 0);        // B*T*D f16     = 8 MB
  half_t* wqh = (half_t*)(ws + 8 * MB);   // D*D f16       = 2 MB
  half_t* wkh = (half_t*)(ws + 10 * MB);
  half_t* wvh = (half_t*)(ws + 12 * MB);
  half_t* woh = (half_t*)(ws + 14 * MB);
  half_t* qh  = (half_t*)(ws + 16 * MB);  // [B,H,T,64]    = 8 MB
  half_t* kh  = (half_t*)(ws + 24 * MB);  // [B,H,T,64]    = 8 MB
  half_t* vth = (half_t*)(ws + 32 * MB);  // [B,H,64,T]    = 8 MB
  half_t* ah  = (half_t*)(ws + 40 * MB);  // [B,T,D]       = 8 MB
  float*  gg  = (float*)(ws + 48 * MB);   // [B,H,T]       = 256 KB

  cvt_f32_to_f16<<<2048, 256, 0, stream>>>(x,  xh,  Bn * Tn * Dn);
  cvt_f32_to_f16<<<1024, 256, 0, stream>>>(Wq, wqh, Dn * Dn);
  cvt_f32_to_f16<<<1024, 256, 0, stream>>>(Wk, wkh, Dn * Dn);
  cvt_f32_to_f16<<<1024, 256, 0, stream>>>(Wv, wvh, Dn * Dn);
  cvt_f32_to_f16<<<1024, 256, 0, stream>>>(Wo, woh, Dn * Dn);

  gates_kernel<<<(Bn * Hn * Tn) / 256, 256, 0, stream>>>(x, Wg, bg, gru, gg);

  qkv_gemm<<<dim3(Mn / 128, Dn / 64, 3), 256, 0, stream>>>(xh, wqh, wkh, wvh,
                                                           bq, bk, bv, qh, kh, vth);

  attn_fused<<<dim3(Hn * (Tn / 128), Bn), 256, 0, stream>>>(qh, kh, vth, pbias, gg, ah);

  out_gemm<<<dim3(Mn / 128, Dn / 64), 256, 0, stream>>>(ah, woh, bo, out);
}
